// FactorGraphGRU_87548613361832
// MI455X (gfx1250) — matrix operand order repaired and checked by reading, not compiled, not run on hardware
//
#include <hip/hip_runtime.h>
#include <cstdint>
#include <cstddef>

#define NN 8192
#define HH 64
#define ALPHA_LRELU 0.2f

typedef __attribute__((ext_vector_type(16))) _Float16       v16h;
typedef __attribute__((ext_vector_type(8)))  float          v8f;
typedef __attribute__((ext_vector_type(4)))  float          f4;
typedef __attribute__((ext_vector_type(8)))  unsigned short us8;
typedef __attribute__((ext_vector_type(16))) unsigned short us16;

// ---------------------------------------------------------------------------
// Load f16 p[0..7] and p[16..23] (stored as u16 bits) -> v16h WMMA B fragment slice.
// ---------------------------------------------------------------------------
__device__ __forceinline__ v16h load_f16x16(const unsigned short* p) {
  us8 lo = *(const us8*)p;        // global_load_b128
  us8 hi = *(const us8*)(p + 16); // global_load_b128
  us16 u;
#pragma unroll
  for (int i = 0; i < 8; ++i) { u[i] = lo[i]; u[i + 8] = hi[i]; }
  return __builtin_bit_cast(v16h, u);
}

__device__ __forceinline__ unsigned short f16bits(float f) {
  _Float16 h = (_Float16)f;
  return __builtin_bit_cast(unsigned short, h);
}

// ---------------------------------------------------------------------------
// Kernel 1: hw = h @ W_gat (f32 VALU, tiny), emit hcatT (f16, [128][N]) where
// rows 0..63 = h^T, rows 64..127 = hw^T; gather adjacency diagonals.
// ---------------------------------------------------------------------------
__global__ __launch_bounds__(64)
void fg_prep(const float* __restrict__ h, const float* __restrict__ W,
             const float* __restrict__ node_adj, const float* __restrict__ edge_adj,
             unsigned short* __restrict__ hcatT,
             float* __restrict__ dnode, float* __restrict__ dedge) {
  const int n = blockIdx.x;
  const int j = threadIdx.x;
  __shared__ float hrow[HH];
  hrow[j] = h[(size_t)n * HH + j];
  __syncthreads();
  float s = 0.0f;
#pragma unroll 8
  for (int k = 0; k < HH; ++k) s += hrow[k] * W[k * HH + j];
  hcatT[(size_t)j * NN + n]        = f16bits(hrow[j]);
  hcatT[(size_t)(HH + j) * NN + n] = f16bits(s);
  if (j == 0) {
    dnode[n] = node_adj[(size_t)n * NN + n];
    dedge[n] = edge_adj[(size_t)n * NN + n];
  }
}

// ---------------------------------------------------------------------------
// One K-step (K=32) of the masked matmuls. DIAG=true only for the single
// 32-aligned K tile that contains the diagonal of this workgroup's rows.
// sgn = +1 (waves 0-3: >0 masks / relu) or -1 (waves 4-7: <0 masks / min0).
// ---------------------------------------------------------------------------
template <bool DIAG>
__device__ __forceinline__ void mm_step(
    int k0, int grow, int hiHalf, float sgn,
    const float* __restrict__ nrow, const float* __restrict__ erow,
    const unsigned short* __restrict__ bp0, const unsigned short* __restrict__ bp1,
    v8f& acc0, v8f& acc1, v8f& acc2, unsigned int& cp) {
  // A operand raw tiles (16x32 f32), per ISA f16 A layout:
  //   lanes 0-15: K = k0+0..7 , k0+16..23 ; lanes 16-31: +8 shift.
  const int kbA = k0 + hiHalf * 8;
  f4 nv0 = *(const f4*)(nrow + kbA);
  f4 nv1 = *(const f4*)(nrow + kbA + 4);
  f4 nv2 = *(const f4*)(nrow + kbA + 16);
  f4 nv3 = *(const f4*)(nrow + kbA + 20);
  f4 ev0 = *(const f4*)(erow + kbA);
  f4 ev1 = *(const f4*)(erow + kbA + 4);
  f4 ev2 = *(const f4*)(erow + kbA + 16);
  f4 ev3 = *(const f4*)(erow + kbA + 20);
  float nva[16], eva[16];
#pragma unroll
  for (int i = 0; i < 4; ++i) {
    nva[i] = nv0[i]; nva[4 + i] = nv1[i]; nva[8 + i] = nv2[i]; nva[12 + i] = nv3[i];
    eva[i] = ev0[i]; eva[4 + i] = ev1[i]; eva[8 + i] = ev2[i]; eva[12 + i] = ev3[i];
  }
  v16h aN, aE;
#pragma unroll
  for (int q = 0; q < 16; ++q) {
    float nv = nva[q];
    float ev = eva[q];
    if (DIAG) {
      const int  k    = kbA + q + (q < 8 ? 0 : 8);  // element q -> global K index
      const bool keep = (k != grow);                // zero the diagonal
      nv = keep ? nv : 0.0f;
      ev = keep ? ev : 0.0f;
    }
    const float te = fmaxf(sgn * ev, 0.0f);         // |edge value| if sign matches
    aN[q] = (sgn * nv > 0.0f) ? (_Float16)1.0f : (_Float16)0.0f;
    aE[q] = (_Float16)(sgn * te);                   // relu(ev) or min(ev,0)
    cp += (te > 0.0f) ? 1u : 0u;                    // branch-free count, every wave
  }
  // B operands (same K order as A: lanes 0-15 K=k0+0..7, k0+16..23 ; lanes 16-31: +8 shift)
  const int koff = k0 + hiHalf * 8;
  v16h b0 = load_f16x16(bp0 + koff);
  v16h b1 = load_f16x16(bp1 + koff);
  acc0 = __builtin_amdgcn_wmma_f32_16x16x32_f16(false, aN, false, b0, (short)0, acc0, false, false);
  acc1 = __builtin_amdgcn_wmma_f32_16x16x32_f16(false, aN, false, b1, (short)0, acc1, false, false);
  acc2 = __builtin_amdgcn_wmma_f32_16x16x32_f16(false, aE, false, b1, (short)0, acc2, false, false);
}

// ---------------------------------------------------------------------------
// Kernel 2: the six masked N x N x 64 matmuls via v_wmma_f32_16x16x32_f16.
// Block = 256 threads = 8 waves; block owns a 16-row output tile.
// Wave w (wq = w&3, neg = w>=4) accumulates:
//   acc0: mask(node) @ hcat cols [16wq .. 16wq+15]       (node_support slice)
//   acc1: mask(node) @ hcat cols [64+16wq .. 64+16wq+15] (h_plus / h_minus)
//   acc2: tf(edge)   @ hw   cols [16wq .. 16wq+15]       (E1 / E2)
// ---------------------------------------------------------------------------
__global__ __launch_bounds__(256)
void fg_masked_matmul(const float* __restrict__ node_adj,
                      const float* __restrict__ edge_adj,
                      const unsigned short* __restrict__ hcatT,
                      float* __restrict__ node_support,
                      float* __restrict__ h_plus, float* __restrict__ h_minus,
                      float* __restrict__ E1, float* __restrict__ E2,
                      unsigned int* __restrict__ cnt_p,
                      unsigned int* __restrict__ cnt_m) {
  const int   wave   = threadIdx.x >> 5;
  const int   lane   = threadIdx.x & 31;
  const int   wq     = wave & 3;
  const bool  neg    = wave >= 4;
  const float sgn    = neg ? -1.0f : 1.0f;
  const int   m0     = blockIdx.x * 16;
  const int   row16  = lane & 15;
  const int   hiHalf = lane >> 4;          // 0: lanes 0-15, 1: lanes 16-31
  const int   grow   = m0 + row16;         // global output row held by this lane

  const float* nrow = node_adj + (size_t)grow * NN;
  const float* erow = edge_adj + (size_t)grow * NN;
  // B fragments: lane = column; hcatT row index = hcat column.
  const unsigned short* bp0 = hcatT + (size_t)(16 * wq + row16) * NN;
  const unsigned short* bp1 = hcatT + (size_t)(HH + 16 * wq + row16) * NN;

  v8f acc0 = {}, acc1 = {}, acc2 = {};
  unsigned int cp = 0;

  // Rows m0..m0+15 share one 32-aligned K block containing the diagonal.
  const int diagK0 = m0 & ~31;
  for (int k0 = 0; k0 < diagK0; k0 += 32)
    mm_step<false>(k0, grow, hiHalf, sgn, nrow, erow, bp0, bp1, acc0, acc1, acc2, cp);
  mm_step<true>(diagK0, grow, hiHalf, sgn, nrow, erow, bp0, bp1, acc0, acc1, acc2, cp);
  for (int k0 = diagK0 + 32; k0 < NN; k0 += 32)
    mm_step<false>(k0, grow, hiHalf, sgn, nrow, erow, bp0, bp1, acc0, acc1, acc2, cp);

  // --- store D fragments (f32 C/D layout: lane = col, VGPR r -> M = 8*hiHalf + r)
  const int orow = m0 + hiHalf * 8;
  const int ocol = lane & 15;
  {
    float* p0 = node_support + (size_t)orow * 128 + (neg ? HH : 0) + 16 * wq + ocol;
#pragma unroll
    for (int r = 0; r < 8; ++r) p0[(size_t)r * 128] = acc0[r];
    float* p1 = (neg ? h_minus : h_plus) + (size_t)orow * HH + 16 * wq + ocol;
#pragma unroll
    for (int r = 0; r < 8; ++r) p1[(size_t)r * HH] = acc1[r];
    float* p2 = (neg ? E2 : E1) + (size_t)orow * HH + 16 * wq + ocol;
#pragma unroll
    for (int r = 0; r < 8; ++r) p2[(size_t)r * HH] = acc2[r];
  }
  // Edge sign counts: every wave counted; only waves 0 / 4 publish.
  if (wave == 0 || wave == 4) {
    unsigned int tot = cp + (unsigned int)__shfl_xor((int)cp, 16, 32);
    if (lane < 16) (wave == 0 ? cnt_p : cnt_m)[m0 + lane] = tot;
  }
}

// ---------------------------------------------------------------------------
// Kernel 3: per-row GAT scalars (leaky-relu scores, 2-value softmax), both
// GRU cells, and the diag-scaled combine. 256 threads = 4 rows x 64 features.
// ---------------------------------------------------------------------------
__global__ __launch_bounds__(256)
void fg_finalize(const float* __restrict__ h, const float* __restrict__ a_gat,
                 const float* __restrict__ w_ih_e, const float* __restrict__ w_hh_e,
                 const float* __restrict__ b_ih_e, const float* __restrict__ b_hh_e,
                 const float* __restrict__ w_ih_n, const float* __restrict__ w_hh_n,
                 const float* __restrict__ b_ih_n, const float* __restrict__ b_hh_n,
                 const float* __restrict__ node_support,
                 const float* __restrict__ h_plus, const float* __restrict__ h_minus,
                 const float* __restrict__ E1, const float* __restrict__ E2,
                 const unsigned int* __restrict__ cnt_p, const unsigned int* __restrict__ cnt_m,
                 const float* __restrict__ dnode, const float* __restrict__ dedge,
                 float* __restrict__ out) {
  const int g   = threadIdx.x >> 6;   // row-group within block (0..3)
  const int j   = threadIdx.x & 63;   // feature
  const int row = blockIdx.x * 4 + g;

  __shared__ float ns[4][128];
  __shared__ float hr[4][HH];
  __shared__ float es[4][HH];
  __shared__ float red[4][HH];
  __shared__ float scal[4][2];

  ns[g][j]      = node_support[(size_t)row * 128 + j];
  ns[g][j + HH] = node_support[(size_t)row * 128 + HH + j];
  hr[g][j]      = h[(size_t)row * HH + j];
  const float hp = h_plus[(size_t)row * HH + j];
  const float hm = h_minus[(size_t)row * HH + j];

  // e_plus pre-activation: concat[h_plus, h_minus] @ a
  red[g][j] = hp * a_gat[j] + hm * a_gat[HH + j];
  __syncthreads();
#pragma unroll
  for (int s = 32; s > 0; s >>= 1) { if (j < s) red[g][j] += red[g][j + s]; __syncthreads(); }
  const float ep_pre = red[g][0];
  __syncthreads();
  // e_minus pre-activation: concat[h_minus, h_plus] @ a
  red[g][j] = hm * a_gat[j] + hp * a_gat[HH + j];
  __syncthreads();
#pragma unroll
  for (int s = 32; s > 0; s >>= 1) { if (j < s) red[g][j] += red[g][j + s]; __syncthreads(); }
  const float em_pre = red[g][0];

  if (j == 0) {
    const float e_p = ep_pre > 0.0f ? ep_pre : ALPHA_LRELU * ep_pre;
    const float e_m = em_pre > 0.0f ? em_pre : ALPHA_LRELU * em_pre;
    const float p = (float)cnt_p[row];
    const float m = (float)cnt_m[row];
    float sp = 0.0f, sm = 0.0f;
    if (p > 0.0f || m > 0.0f) {
      float M = -3.0e38f;
      if (p > 0.0f) M = e_p;
      if (m > 0.0f) M = fmaxf(M, e_m);
      const float wp  = (p > 0.0f) ? __expf(e_p - M) : 0.0f;
      const float wm  = (m > 0.0f) ? __expf(e_m - M) : 0.0f;
      const float den = p * wp + m * wm;
      sp = wp / den; sm = wm / den;
    }
    scal[g][0] = sp; scal[g][1] = sm;
  }
  __syncthreads();
  const float sp = scal[g][0], sm = scal[g][1];
  es[g][j] = sp * E1[(size_t)row * HH + j] + sm * E2[(size_t)row * HH + j];
  __syncthreads();

  // ---- edge GRU: x = node_support (128), hidden = h ----
  float gir = b_ih_e[j], giz = b_ih_e[HH + j], gin = b_ih_e[128 + j];
  {
    const float* wr = w_ih_e + (size_t)j * 128;
    const float* wz = w_ih_e + (size_t)(HH + j) * 128;
    const float* wn = w_ih_e + (size_t)(128 + j) * 128;
#pragma unroll 4
    for (int k = 0; k < 128; ++k) { const float x = ns[g][k]; gir += x * wr[k]; giz += x * wz[k]; gin += x * wn[k]; }
  }
  float ghr = b_hh_e[j], ghz = b_hh_e[HH + j], ghn = b_hh_e[128 + j];
  {
    const float* ur = w_hh_e + (size_t)j * HH;
    const float* uz = w_hh_e + (size_t)(HH + j) * HH;
    const float* un = w_hh_e + (size_t)(128 + j) * HH;
#pragma unroll 4
    for (int k = 0; k < HH; ++k) { const float x = hr[g][k]; ghr += x * ur[k]; ghz += x * uz[k]; ghn += x * un[k]; }
  }
  float rg = 1.0f / (1.0f + __expf(-(gir + ghr)));
  float zg = 1.0f / (1.0f + __expf(-(giz + ghz)));
  float ng = tanhf(gin + rg * ghn);
  const float eo = (1.0f - zg) * ng + zg * hr[g][j];

  // ---- node GRU: x = edge_support (64), hidden = h ----
  float nir = b_ih_n[j], niz = b_ih_n[HH + j], nin = b_ih_n[128 + j];
  {
    const float* wr = w_ih_n + (size_t)j * HH;
    const float* wz = w_ih_n + (size_t)(HH + j) * HH;
    const float* wn = w_ih_n + (size_t)(128 + j) * HH;
#pragma unroll 4
    for (int k = 0; k < HH; ++k) { const float x = es[g][k]; nir += x * wr[k]; niz += x * wz[k]; nin += x * wn[k]; }
  }
  float nhr = b_hh_n[j], nhz = b_hh_n[HH + j], nhn = b_hh_n[128 + j];
  {
    const float* ur = w_hh_n + (size_t)j * HH;
    const float* uz = w_hh_n + (size_t)(HH + j) * HH;
    const float* un = w_hh_n + (size_t)(128 + j) * HH;
#pragma unroll 4
    for (int k = 0; k < HH; ++k) { const float x = hr[g][k]; nhr += x * ur[k]; nhz += x * uz[k]; nhn += x * un[k]; }
  }
  rg = 1.0f / (1.0f + __expf(-(nir + nhr)));
  zg = 1.0f / (1.0f + __expf(-(niz + nhz)));
  ng = tanhf(nin + rg * nhn);
  const float no = (1.0f - zg) * ng + zg * hr[g][j];

  out[(size_t)row * HH + j] = dedge[row] * eo + dnode[row] * no;
}

// ---------------------------------------------------------------------------
extern "C" void kernel_launch(void* const* d_in, const int* in_sizes, int n_in,
                              void* d_out, int out_size, void* d_ws, size_t ws_size,
                              hipStream_t stream) {
  (void)in_sizes; (void)n_in; (void)out_size; (void)ws_size;
  const float* h        = (const float*)d_in[0];
  const float* node_adj = (const float*)d_in[1];
  const float* edge_adj = (const float*)d_in[2];
  const float* W_gat    = (const float*)d_in[3];
  const float* a_gat    = (const float*)d_in[4];
  const float* w_ih_e   = (const float*)d_in[5];
  const float* w_hh_e   = (const float*)d_in[6];
  const float* b_ih_e   = (const float*)d_in[7];
  const float* b_hh_e   = (const float*)d_in[8];
  const float* w_ih_n   = (const float*)d_in[9];
  const float* w_hh_n   = (const float*)d_in[10];
  const float* b_ih_n   = (const float*)d_in[11];
  const float* b_hh_n   = (const float*)d_in[12];

  char*  ws  = (char*)d_ws;
  size_t off = 0;
  auto alloc = [&](size_t bytes) -> void* {
    void* p = ws + off;
    off = (off + bytes + 255) & ~(size_t)255;
    return p;
  };
  unsigned short* hcatT        = (unsigned short*)alloc((size_t)128 * NN * 2);  // 2 MB, L2-hot B operand
  float*          node_support = (float*)alloc((size_t)NN * 128 * 4);           // 4 MB
  float*          h_plus       = (float*)alloc((size_t)NN * HH * 4);            // 2 MB
  float*          h_minus      = (float*)alloc((size_t)NN * HH * 4);
  float*          E1           = (float*)alloc((size_t)NN * HH * 4);
  float*          E2           = (float*)alloc((size_t)NN * HH * 4);
  unsigned int*   cnt_p        = (unsigned int*)alloc((size_t)NN * 4);
  unsigned int*   cnt_m        = (unsigned int*)alloc((size_t)NN * 4);
  float*          dnode        = (float*)alloc((size_t)NN * 4);
  float*          dedge        = (float*)alloc((size_t)NN * 4);

  fg_prep<<<NN, 64, 0, stream>>>(h, W_gat, node_adj, edge_adj, hcatT, dnode, dedge);
  fg_masked_matmul<<<NN / 16, 256, 0, stream>>>(node_adj, edge_adj, hcatT, node_support,
                                                h_plus, h_minus, E1, E2, cnt_p, cnt_m);
  fg_finalize<<<NN / 4, 256, 0, stream>>>(h, a_gat, w_ih_e, w_hh_e, b_ih_e, b_hh_e,
                                          w_ih_n, w_hh_n, b_ih_n, b_hh_n,
                                          node_support, h_plus, h_minus, E1, E2,
                                          cnt_p, cnt_m, dnode, dedge, (float*)d_out);
}
